// CascadeQuadtreeAttention_21620865368128
// MI455X (gfx1250) — matrix operand order repaired and checked by reading, not compiled, not run on hardware
//
#include <hip/hip_runtime.h>
#include <hip/hip_bf16.h>

// ---------------------------------------------------------------------------
// CascadeQuadtreeAttention for MI455X (gfx1250, wave32, WMMA)
//
// Shapes (fixed by the reference setup):
//   B=4, H=W=H1=W1=128, C=128, nh=8, hd=16, K=8
//   N = N1 = 16384, Hc=Wc=W1c=64, Ncq = 4096, KV per cell = K*4 = 32
// ---------------------------------------------------------------------------

#define B_      4
#define HW_     128     // H == W == H1 == W1
#define C_      128
#define N_      16384   // H*W
#define NH_     8
#define HD_     16
#define WC_     64      // W/2
#define W1C_    64      // W1/2
#define NCQ_    4096    // Hc*Wc
#define KSEL_   8
#define KV_     32      // K*4
#define M_TOK   (B_ * N_)       // 65536 rows for the dense GEMMs
#define NTILES  (C_ / 16)       // 8
#define KCHUNKS (C_ / 32)       // 4
#define WFRAG_ELEMS (NTILES * KCHUNKS * 32 * 16)   // 16384 bf16 per packed matrix

typedef __attribute__((ext_vector_type(16))) __bf16 v16bf;
typedef __attribute__((ext_vector_type(8)))  float  v8f;

// ---------------------------------------------------------------------------
// Pack one f32 weight matrix [128(out),128(in)] into lane-ordered bf16 WMMA
// B fragments:  frag[tn][kk][lane][e] = bf16( Wt[tn*16 + (lane&15), kb + e] )
// with kb = kk*32 + (lane>>4)*16  (contiguous k = 16h + e within each 32-k chunk, wave32).
// grid = (8 tn, 4 kk), block = 32.
// ---------------------------------------------------------------------------
__global__ void pack_weights_bf16(const float* __restrict__ Wt,
                                  __bf16* __restrict__ out)
{
    const int tn   = blockIdx.x;
    const int kk   = blockIdx.y;
    const int lane = threadIdx.x;
    const int ncol = tn * 16 + (lane & 15);
    const int kb   = kk * 32 + (lane >> 4) * 16;
    __bf16* o = out + (size_t)(((tn * KCHUNKS + kk) * 32) + lane) * 16;
    #pragma unroll
    for (int e = 0; e < 16; ++e)
        o[e] = (__bf16)Wt[(size_t)ncol * C_ + kb + e];
}

// ---------------------------------------------------------------------------
// Dense GEMM:  Out[m, n] = sum_c In[m, c] * Wt[n, c]  (+ bias[n])
// One wave computes a full 16x128 output strip: 8 f32 accumulators,
// 32 x v_wmma_f32_16x16x32_bf16 per wave. A fragment converted once per
// K-chunk and reused across all 8 N tiles; B fragments are pre-packed bf16
// loaded straight into the WMMA operand layout (no conversion VALU).
// grid = M/16, block = 32 (one wave, EXEC all ones for WMMA).
// ---------------------------------------------------------------------------
__global__ void gemm_wmma_bf16(const float*  __restrict__ In,
                               const __bf16* __restrict__ Wbf,
                               const float*  __restrict__ bias,
                               float* __restrict__ Out)
{
    const int tm    = blockIdx.x;
    const int lane  = threadIdx.x;
    const int l15   = lane & 15;
    const int khalf = lane >> 4;          // 0 or 1

    const int mrow = tm * 16 + l15;       // A-matrix row held by this lane
    const float* arow = In + (size_t)mrow * C_;

    v8f acc[NTILES] = {};

    #pragma unroll
    for (int kk = 0; kk < KCHUNKS; ++kk) {
        // --- A fragment: 16x32 bf16, lane = M-row (ISA layout) ----------
        v16bf a;
        #pragma unroll
        for (int i = 0; i < 8; ++i) {
            const int k0 = kk * 32 + khalf * 16 + i * 2;
            a[2 * i]     = (__bf16)arow[k0];
            a[2 * i + 1] = (__bf16)arow[k0 + 1];
        }
        // --- sweep all 8 N tiles with this A fragment -------------------
        #pragma unroll
        for (int tn = 0; tn < NTILES; ++tn) {
            const v16bf b = *(const v16bf*)(
                Wbf + (size_t)(((tn * KCHUNKS + kk) * 32) + lane) * 16);
            acc[tn] = __builtin_amdgcn_wmma_f32_16x16x32_bf16(
                          /*neg_a=*/false, a, /*neg_b=*/false, b,
                          /*c_mod=*/(short)0, acc[tn],
                          /*reuse_a=*/false, /*reuse_b=*/false);
        }
    }

    // --- store C/D: lane holds N = ncol, rows M = mbase + j --------------
    const int mbase = tm * 16 + khalf * 8;
    #pragma unroll
    for (int tn = 0; tn < NTILES; ++tn) {
        const int ncol = tn * 16 + l15;
        const float bv = (bias != nullptr) ? bias[ncol] : 0.0f;
        #pragma unroll
        for (int j = 0; j < 8; ++j)
            Out[(size_t)(mbase + j) * C_ + ncol] = acc[tn][j] + bv;
    }
}

// ---------------------------------------------------------------------------
// Gathered quadtree attention.
// One wave per (b, h, coarse cell). lane = kj (0..31 gathered KV rows).
//   scores[qi, kj] = <q[qi,:], kg[kj,:]> * 0.25 + rel_pos[..., qi, kj]
//   softmax over kj (wave-wide shfl reductions, wave32)
//   out[qi, d] = sum_kj attn[qi,kj] * vg[kj, d]   (via LDS staging)
// grid = (Ncq, nh, B), block = 32.
// ---------------------------------------------------------------------------
__global__ void attn_kernel(const float* __restrict__ q,
                            const float* __restrict__ k,
                            const float* __restrict__ v,
                            const int*   __restrict__ idx,
                            const float* __restrict__ rel_pos,
                            float* __restrict__ msg)
{
    const int nc   = blockIdx.x;   // coarse cell
    const int h    = blockIdx.y;   // head
    const int b    = blockIdx.z;   // batch
    const int lane = threadIdx.x;  // == kj

    __shared__ float qs[4 * HD_];
    __shared__ float vs[KV_ * HD_];
    __shared__ float attn_s[4 * KV_];

    // ---- gather this lane's KV row ------------------------------------
    const int kk    = lane >> 2;
    const int child = lane & 3;
    const int m     = idx[((size_t)b * NCQ_ + nc) * KSEL_ + kk];
    const int ci    = m / W1C_;
    const int cj    = m % W1C_;
    const int tok   = (2 * ci + (child >> 1)) * HW_ + (2 * cj + (child & 1));

    const float* kp = k + ((size_t)b * N_ + tok) * C_ + h * HD_;
    const float* vp = v + ((size_t)b * N_ + tok) * C_ + h * HD_;

    float kreg[HD_];
    #pragma unroll
    for (int d = 0; d < HD_; ++d) {
        kreg[d] = kp[d];
        vs[lane * HD_ + d] = vp[d];
    }

    // ---- load the 4 fine queries of this cell -------------------------
    const int ciq = nc / WC_;
    const int cjq = nc % WC_;
    if (lane < 4) {
        const int qtok = (2 * ciq + (lane >> 1)) * HW_ + (2 * cjq + (lane & 1));
        const float* qp = q + ((size_t)b * N_ + qtok) * C_ + h * HD_;
        #pragma unroll
        for (int d = 0; d < HD_; ++d) qs[lane * HD_ + d] = qp[d];
    }
    __syncthreads();

    // ---- scores + softmax ---------------------------------------------
    const float scale = 0.25f;  // hd^-0.5, hd = 16
    const float* rp = rel_pos +
        ((((size_t)b * NH_ + h) * NCQ_ + nc) * 4) * KV_ + lane;

    #pragma unroll
    for (int qi = 0; qi < 4; ++qi) {
        float s = 0.0f;
        #pragma unroll
        for (int d = 0; d < HD_; ++d) s += qs[qi * HD_ + d] * kreg[d];
        s = s * scale + rp[qi * KV_];

        float mx = s;
        #pragma unroll
        for (int off = 16; off >= 1; off >>= 1)
            mx = fmaxf(mx, __shfl_xor(mx, off, 32));
        const float e = __expf(s - mx);
        float sum = e;
        #pragma unroll
        for (int off = 16; off >= 1; off >>= 1)
            sum += __shfl_xor(sum, off, 32);
        attn_s[qi * KV_ + lane] = e / sum;
    }
    __syncthreads();

    // ---- attn @ V : each lane writes 2 output elements ------------------
    const int d = lane & 15;
    #pragma unroll
    for (int r = 0; r < 2; ++r) {
        const int qi = 2 * (lane >> 4) + r;
        float acc = 0.0f;
        #pragma unroll
        for (int kj = 0; kj < KV_; ++kj)
            acc += attn_s[qi * KV_ + kj] * vs[kj * HD_ + d];
        const int qtok = (2 * ciq + (qi >> 1)) * HW_ + (2 * cjq + (qi & 1));
        msg[((size_t)b * N_ + qtok) * C_ + h * HD_ + d] = acc;
    }
}

// ---------------------------------------------------------------------------
// Upsampled index output: [B, Ncq, K*4] int32
// ---------------------------------------------------------------------------
__global__ void upsample_idx_kernel(const int* __restrict__ idx,
                                    int* __restrict__ out,
                                    int total /* B*Ncq*K*4 */)
{
    const int i = blockIdx.x * blockDim.x + threadIdx.x;
    if (i >= total) return;
    const int child = i & 3;
    const int m = idx[i >> 2];
    const int ci = m / W1C_;
    const int cj = m % W1C_;
    out[i] = (2 * ci + (child >> 1)) * HW_ + (2 * cj + (child & 1));
}

// ---------------------------------------------------------------------------
extern "C" void kernel_launch(void* const* d_in, const int* in_sizes, int n_in,
                              void* d_out, int out_size, void* d_ws, size_t ws_size,
                              hipStream_t stream)
{
    (void)in_sizes; (void)n_in; (void)out_size; (void)ws_size;

    // setup_inputs order: x, target, H, W, H1, W1, idx, rel_pos, Wq, Wk, Wv, Wo, bo
    const float* x       = (const float*)d_in[0];
    const float* target  = (const float*)d_in[1];
    const int*   idx     = (const int*)  d_in[6];
    const float* rel_pos = (const float*)d_in[7];
    const float* Wq      = (const float*)d_in[8];
    const float* Wk      = (const float*)d_in[9];
    const float* Wv      = (const float*)d_in[10];
    const float* Wo      = (const float*)d_in[11];
    const float* bo      = (const float*)d_in[12];

    const size_t tok_elems = (size_t)M_TOK * C_;   // 8,388,608 floats per buffer
    float* q   = (float*)d_ws;
    float* k   = q   + tok_elems;
    float* v   = k   + tok_elems;
    float* msg = v   + tok_elems;
    __bf16* wq_bf = (__bf16*)(msg + tok_elems);
    __bf16* wk_bf = wq_bf + WFRAG_ELEMS;
    __bf16* wv_bf = wk_bf + WFRAG_ELEMS;
    __bf16* wo_bf = wv_bf + WFRAG_ELEMS;

    float* x_out   = (float*)d_out;
    int*   idx_out = (int*)(x_out + tok_elems);

    const dim3 wave(32, 1, 1);

    // 0) pack the four weight matrices into bf16 WMMA B fragments
    const dim3 pgrid(NTILES, KCHUNKS, 1);
    pack_weights_bf16<<<pgrid, wave, 0, stream>>>(Wq, wq_bf);
    pack_weights_bf16<<<pgrid, wave, 0, stream>>>(Wk, wk_bf);
    pack_weights_bf16<<<pgrid, wave, 0, stream>>>(Wv, wv_bf);
    pack_weights_bf16<<<pgrid, wave, 0, stream>>>(Wo, wo_bf);

    // 1) projections: q = x@Wq^T, k = target@Wk^T, v = target@Wv^T
    const dim3 ggrid(M_TOK / 16, 1, 1);   // 4096 strips of 16x128
    gemm_wmma_bf16<<<ggrid, wave, 0, stream>>>(x,      wq_bf, nullptr, q);
    gemm_wmma_bf16<<<ggrid, wave, 0, stream>>>(target, wk_bf, nullptr, k);
    gemm_wmma_bf16<<<ggrid, wave, 0, stream>>>(target, wv_bf, nullptr, v);

    // 2) gathered quadtree attention -> msg
    attn_kernel<<<dim3(NCQ_, NH_, B_), wave, 0, stream>>>(q, k, v, idx, rel_pos, msg);

    // 3) output projection: x_out = msg@Wo^T + bo
    gemm_wmma_bf16<<<ggrid, wave, 0, stream>>>(msg, wo_bf, bo, x_out);

    // 4) upsampled indices
    const int total_idx = B_ * NCQ_ * KSEL_ * 4;   // 524288
    upsample_idx_kernel<<<(total_idx + 255) / 256, 256, 0, stream>>>(idx, idx_out, total_idx);
}
